// CrossAttention_28845000360289
// MI455X (gfx1250) — matrix operand order repaired and checked by reading, not compiled, not run on hardware
//
#include <hip/hip_runtime.h>

#define HEADS 16
#define DIM_HEAD 64
#define QDIM 1024
#define INNER 1024
#define BATCH 2
#define SEQ_N 2048
#define SEQ_M 2048
#define ATT_SCALE 0.125f  // 64^-0.5, folded into Wq conversion

typedef __attribute__((ext_vector_type(16))) __bf16 v16bf;
typedef __attribute__((ext_vector_type(8)))  float  v8f;

// ---------- helpers ----------

__device__ __forceinline__ unsigned short f2bf(float f) {
  union { float f; unsigned int u; } x; x.f = f;
  unsigned int u = x.u;
  return (unsigned short)((u + 0x7FFFu + ((u >> 16) & 1u)) >> 16);  // RNE
}

__device__ __forceinline__ v8f v8f_zero() {
  v8f z;
#pragma unroll
  for (int i = 0; i < 8; ++i) z[i] = 0.0f;
  return z;
}

__device__ __forceinline__ v8f wmma_bf16(v16bf a, v16bf b, v8f c) {
  // D = A(16x32 bf16) * B(32x16 bf16) + C(16x16 f32)
  return __builtin_amdgcn_wmma_f32_16x16x32_bf16(false, a, false, b,
                                                 (short)0, c, false, false);
}

// Load a 16x32 A-fragment or 32x16 B-fragment from an LDS row that holds
// K contiguously (rowptr points at k=0 of this lane's M/N row).
// A layout (ISA 7.12.2): lane<16: vgpr0-3 K=0..7, vgpr4-7 K=16..23; lane>=16 +8.
// B layout: same K order as A (lane<16: vgpr0-3 K=0..7, vgpr4-7 K=16..23; lane>=16 +8).
template <bool IS_A>
__device__ __forceinline__ v16bf load_frag(const unsigned short* rowptr,
                                           int kbase, int half) {
  union { v16bf v; unsigned int p[8]; } u;
#pragma unroll
  for (int i = 0; i < 8; ++i) {
    int kb = ((i & 4) ? 16 : 0) + half * 8 + (i & 3) * 2;
    u.p[i] = *reinterpret_cast<const unsigned int*>(rowptr + kbase + kb);
  }
  return u.v;
}

__device__ __forceinline__ void store_out(float* p, float v) { *p = v; }
__device__ __forceinline__ void store_out(unsigned short* p, float v) { *p = f2bf(v); }

// ---------- fp32 -> bf16 convert (with optional static scale fold) ----------

__global__ void f32_to_bf16_kernel(const float* __restrict__ in,
                                   unsigned short* __restrict__ out,
                                   int n4, float scale) {
  int i = blockIdx.x * blockDim.x + threadIdx.x;
  if (i < n4) {
    float4 f = ((const float4*)in)[i];
    ushort4 o;
    o.x = f2bf(f.x * scale); o.y = f2bf(f.y * scale);
    o.z = f2bf(f.z * scale); o.w = f2bf(f.w * scale);
    ((ushort4*)out)[i] = o;
  }
}

// ---------- bf16 GEMM: C[M,N] = A[M,K] * B[K,N] ----------
// 128x128 block, BK=32, double-buffered LDS, register prefetch of next tile.

template <typename OutT>
__global__ void __launch_bounds__(256)
gemm_bf16_128x128(const unsigned short* __restrict__ A,
                  const unsigned short* __restrict__ Bm,
                  OutT* __restrict__ C, int M, int N, int K) {
  __shared__ unsigned short sA[2][128 * 40];   // [m][k], stride 40
  __shared__ unsigned short sBt[2][128 * 40];  // [n][k], transposed, stride 40

  const int bm = blockIdx.y * 128, bn = blockIdx.x * 128;
  const int t = threadIdx.x;
  const int lane = t & 31, w = t >> 5;
  const int wm = w >> 2, wn = w & 3;            // 2x4 wave grid -> 64x32 per wave
  const int half = (lane >> 4) & 1, ln = lane & 15;

  // staging assignments
  const int am = t >> 1, aseg = (t & 1) * 16;          // A: row, 16-elem k-seg
  const int bk = t >> 3, bnseg = (t & 7) * 16;         // B: k-row, 16-elem n-seg
  const unsigned short* aptr = A + (size_t)(bm + am) * K + aseg;
  const unsigned short* bptr = Bm + (size_t)bk * N + bn + bnseg;

  v8f acc[4][2];
#pragma unroll
  for (int mt = 0; mt < 4; ++mt)
#pragma unroll
    for (int nt = 0; nt < 2; ++nt) acc[mt][nt] = v8f_zero();

  uint4 ar0, ar1;
  union { uint4 q[2]; unsigned short u[16]; } br;

  // prologue: tile 0 -> buffer 0
  ar0 = ((const uint4*)aptr)[0]; ar1 = ((const uint4*)aptr)[1];
  br.q[0] = ((const uint4*)bptr)[0]; br.q[1] = ((const uint4*)bptr)[1];
  {
    uint4* dst = (uint4*)(&sA[0][am * 40 + aseg]);
    dst[0] = ar0; dst[1] = ar1;
#pragma unroll
    for (int i = 0; i < 16; ++i) sBt[0][(bnseg + i) * 40 + bk] = br.u[i];
  }
  __syncthreads();

  const int steps = K / 32;
  for (int it = 0; it < steps; ++it) {
    const int buf = it & 1;
    // prefetch next tile into registers (overlaps with WMMA below)
    if (it + 1 < steps) {
      const unsigned short* ap = aptr + (it + 1) * 32;
      ar0 = ((const uint4*)ap)[0]; ar1 = ((const uint4*)ap)[1];
      const unsigned short* bp = bptr + (size_t)(it + 1) * 32 * N;
      br.q[0] = ((const uint4*)bp)[0]; br.q[1] = ((const uint4*)bp)[1];
    }

    v16bf afr[4], bfr[2];
#pragma unroll
    for (int mt = 0; mt < 4; ++mt)
      afr[mt] = load_frag<true>(&sA[buf][(wm * 64 + mt * 16 + ln) * 40], 0, half);
#pragma unroll
    for (int nt = 0; nt < 2; ++nt)
      bfr[nt] = load_frag<false>(&sBt[buf][(wn * 32 + nt * 16 + ln) * 40], 0, half);
#pragma unroll
    for (int mt = 0; mt < 4; ++mt)
#pragma unroll
      for (int nt = 0; nt < 2; ++nt)
        acc[mt][nt] = wmma_bf16(afr[mt], bfr[nt], acc[mt][nt]);

    // commit prefetched tile to the other buffer
    if (it + 1 < steps) {
      uint4* dst = (uint4*)(&sA[buf ^ 1][am * 40 + aseg]);
      dst[0] = ar0; dst[1] = ar1;
#pragma unroll
      for (int i = 0; i < 16; ++i) sBt[buf ^ 1][(bnseg + i) * 40 + bk] = br.u[i];
    }
    __syncthreads();
  }

  // C/D layout: value (m = r + 8*half, n = ln) in vgpr r
#pragma unroll
  for (int mt = 0; mt < 4; ++mt)
#pragma unroll
    for (int nt = 0; nt < 2; ++nt)
#pragma unroll
      for (int r = 0; r < 8; ++r) {
        int row = bm + wm * 64 + mt * 16 + r + 8 * half;
        int col = bn + wn * 32 + nt * 16 + ln;
        store_out(C + (size_t)row * N + col, acc[mt][nt][r]);
      }
}

// ---------- flash attention: one block = (b, h, 128 query rows) ----------
// Double-buffered K/V tiles (64 keys), register prefetch, Q cached in regs.

__global__ void __launch_bounds__(256)
flash_attn_kernel(const unsigned short* __restrict__ Q,
                  const unsigned short* __restrict__ Kb,
                  const unsigned short* __restrict__ Vb,
                  unsigned short* __restrict__ Ob) {
  __shared__ unsigned short smem[27648];    // 54 KB
  unsigned short* sQP = smem;               // 128 x 72 : Q tile, reused as P tile
  unsigned short* sK  = smem + 9216;        // 2 x (64 x 72) : [key][d]
  unsigned short* sVt = smem + 18432;       // 2 x (64 x 72) : [d][key]

  const int t = threadIdx.x;
  const int lane = t & 31, w = t >> 5;
  const int half = (lane >> 4) & 1, ln = lane & 15;

  const int tiles_n = SEQ_N / 128;
  int nb = blockIdx.x % tiles_n;
  int bh = blockIdx.x / tiles_n;
  int h = bh % HEADS, b = bh / HEADS;
  const int rowBase = nb * 128;

  const unsigned short* Qg = Q  + ((size_t)b * SEQ_N) * INNER + h * DIM_HEAD;
  const unsigned short* Kg = Kb + ((size_t)b * SEQ_M) * INNER + h * DIM_HEAD;
  const unsigned short* Vg = Vb + ((size_t)b * SEQ_M) * INNER + h * DIM_HEAD;

  // staging assignments for K/V: 64 keys x 64 d, 16 elems per thread-chunk
  const int skey = t >> 2, sseg = (t & 3) * 16;

  {  // stage Q tile 128x64
    int row = t >> 1, seg = (t & 1) * 32;
    const uint4* src = (const uint4*)(Qg + (size_t)(rowBase + row) * INNER + seg);
    uint4* dst = (uint4*)(sQP + row * 72 + seg);
#pragma unroll
    for (int i = 0; i < 4; ++i) dst[i] = src[i];
  }
  {  // prologue: K/V tile 0 -> buffer 0
    const uint4* ks = (const uint4*)(Kg + (size_t)skey * INNER + sseg);
    uint4 k0 = ks[0], k1 = ks[1];
    uint4* kd = (uint4*)(sK + skey * 72 + sseg);
    kd[0] = k0; kd[1] = k1;
    union { uint4 q[2]; unsigned short u[16]; } vr;
    const uint4* vs = (const uint4*)(Vg + (size_t)skey * INNER + sseg);
    vr.q[0] = vs[0]; vr.q[1] = vs[1];
#pragma unroll
    for (int i = 0; i < 16; ++i) sVt[(sseg + i) * 72 + skey] = vr.u[i];
  }
  __syncthreads();

  v16bf qf[2];  // this wave's 16x64 Q block as two 16x32 A fragments
#pragma unroll
  for (int ks = 0; ks < 2; ++ks)
    qf[ks] = load_frag<true>(sQP + (w * 16 + ln) * 72, ks * 32, half);
  asm volatile("s_wait_dscnt 0" ::: "memory");  // Q reads done before sQP reuse
  __syncthreads();

  v8f o[4];
#pragma unroll
  for (int j = 0; j < 4; ++j) o[j] = v8f_zero();
  float m_i[8], l_i[8];
#pragma unroll
  for (int r = 0; r < 8; ++r) { m_i[r] = -3.0e38f; l_i[r] = 0.0f; }

  const int steps = SEQ_M / 64;
  for (int it = 0; it < steps; ++it) {
    const int buf = it & 1;
    const unsigned short* sKb = sK + buf * 4608;
    const unsigned short* sVb = sVt + buf * 4608;

    // prefetch next K/V tile into registers (overlaps softmax + WMMA)
    uint4 kr0, kr1;
    union { uint4 q[2]; unsigned short u[16]; } vr;
    if (it + 1 < steps) {
      const uint4* ksrc =
          (const uint4*)(Kg + (size_t)((it + 1) * 64 + skey) * INNER + sseg);
      kr0 = ksrc[0]; kr1 = ksrc[1];
      const uint4* vsrc =
          (const uint4*)(Vg + (size_t)((it + 1) * 64 + skey) * INNER + sseg);
      vr.q[0] = vsrc[0]; vr.q[1] = vsrc[1];
    }

    // S(16x64) = Qs(16x64) * K^T : B fragment row n = key, k = head dim
    // (ATT_SCALE pre-folded into Wq so S is already scaled)
    v8f s[4];
#pragma unroll
    for (int j = 0; j < 4; ++j) s[j] = v8f_zero();
#pragma unroll
    for (int ks = 0; ks < 2; ++ks)
#pragma unroll
      for (int j = 0; j < 4; ++j) {
        v16bf bf = load_frag<false>(sKb + (j * 16 + ln) * 72, ks * 32, half);
        s[j] = wmma_bf16(qf[ks], bf, s[j]);
      }

    // ---- online softmax; row m = r + 8*half lives in vgpr r across a 16-lane
    // half.  Stage-major xor-reductions: 8 independent bpermutes per stage.
    float mx[8];
#pragma unroll
    for (int r = 0; r < 8; ++r)
      mx[r] = fmaxf(fmaxf(s[0][r], s[1][r]), fmaxf(s[2][r], s[3][r]));
#pragma unroll
    for (int off = 1; off < 16; off <<= 1)
#pragma unroll
      for (int r = 0; r < 8; ++r)
        mx[r] = fmaxf(mx[r], __shfl_xor(mx[r], off, 32));

    float alpha[8];
#pragma unroll
    for (int r = 0; r < 8; ++r) {
      float mn = fmaxf(m_i[r], mx[r]);
      alpha[r] = __expf(m_i[r] - mn);
      m_i[r] = mn;
    }
#pragma unroll
    for (int j = 0; j < 4; ++j)
#pragma unroll
      for (int r = 0; r < 8; ++r) s[j][r] = __expf(s[j][r] - m_i[r]);

    float sm[8];
#pragma unroll
    for (int r = 0; r < 8; ++r)
      sm[r] = (s[0][r] + s[1][r]) + (s[2][r] + s[3][r]);
#pragma unroll
    for (int off = 1; off < 16; off <<= 1)
#pragma unroll
      for (int r = 0; r < 8; ++r) sm[r] += __shfl_xor(sm[r], off, 32);
#pragma unroll
    for (int r = 0; r < 8; ++r) l_i[r] = l_i[r] * alpha[r] + sm[r];
#pragma unroll
    for (int j = 0; j < 4; ++j)
#pragma unroll
      for (int r = 0; r < 8; ++r) o[j][r] *= alpha[r];

    // D-layout -> A-layout reshape of P through this wave's private sQP region
#pragma unroll
    for (int j = 0; j < 4; ++j)
#pragma unroll
      for (int r = 0; r < 8; ++r)
        sQP[(w * 16 + r + 8 * half) * 72 + j * 16 + ln] = f2bf(s[j][r]);

    // O(16x64) += P(16x64) * V(64x64) : B fragment row n = d, k = key
#pragma unroll
    for (int ks = 0; ks < 2; ++ks) {
      v16bf pa = load_frag<true>(sQP + (w * 16 + ln) * 72, ks * 32, half);
#pragma unroll
      for (int j = 0; j < 4; ++j) {
        v16bf vb = load_frag<false>(sVb + (j * 16 + ln) * 72, ks * 32, half);
        o[j] = wmma_bf16(pa, vb, o[j]);
      }
    }

    // commit prefetched K/V to the other buffer
    if (it + 1 < steps) {
      unsigned short* nK = sK + (buf ^ 1) * 4608;
      unsigned short* nV = sVt + (buf ^ 1) * 4608;
      uint4* kd = (uint4*)(nK + skey * 72 + sseg);
      kd[0] = kr0; kd[1] = kr1;
#pragma unroll
      for (int i = 0; i < 16; ++i) nV[(sseg + i) * 72 + skey] = vr.u[i];
    }
    __syncthreads();
  }

  // normalize and store bf16 attention output [b, n, h*64 + d]
#pragma unroll
  for (int j = 0; j < 4; ++j)
#pragma unroll
    for (int r = 0; r < 8; ++r) {
      int row = rowBase + w * 16 + r + 8 * half;
      int col = h * DIM_HEAD + j * 16 + ln;
      Ob[((size_t)b * SEQ_N + row) * INNER + col] = f2bf(o[j][r] / l_i[r]);
    }
}

// ---------- launcher ----------

extern "C" void kernel_launch(void* const* d_in, const int* in_sizes, int n_in,
                              void* d_out, int out_size, void* d_ws, size_t ws_size,
                              hipStream_t stream) {
  (void)in_sizes; (void)n_in; (void)out_size; (void)ws_size;
  const float* x   = (const float*)d_in[0];
  const float* ctx = (const float*)d_in[1];
  const float* Wq  = (const float*)d_in[2];
  const float* Wk  = (const float*)d_in[3];
  const float* Wv  = (const float*)d_in[4];
  const float* Wo  = (const float*)d_in[5];
  float* out = (float*)d_out;

  char* ws = (char*)d_ws;
  const size_t SEQ_B = (size_t)BATCH * SEQ_N * INNER * 2;  // 8 MiB bf16 tensor
  const size_t W_B   = (size_t)QDIM * INNER * 2;           // 2 MiB bf16 weight
  unsigned short* xb  = (unsigned short*)(ws);
  unsigned short* cb  = (unsigned short*)(ws + SEQ_B);
  unsigned short* wqb = (unsigned short*)(ws + 2 * SEQ_B);
  unsigned short* wkb = (unsigned short*)(ws + 2 * SEQ_B + 1 * W_B);
  unsigned short* wvb = (unsigned short*)(ws + 2 * SEQ_B + 2 * W_B);
  unsigned short* wob = (unsigned short*)(ws + 2 * SEQ_B + 3 * W_B);
  unsigned short* qb  = (unsigned short*)(ws + 2 * SEQ_B + 4 * W_B);
  unsigned short* kb  = (unsigned short*)(ws + 3 * SEQ_B + 4 * W_B);
  unsigned short* vb  = (unsigned short*)(ws + 4 * SEQ_B + 4 * W_B);
  unsigned short* ab  = (unsigned short*)(ws + 5 * SEQ_B + 4 * W_B);

  const int n4seq = BATCH * SEQ_N * QDIM / 4;
  const int n4w   = QDIM * INNER / 4;
  f32_to_bf16_kernel<<<(n4seq + 255) / 256, 256, 0, stream>>>(x,   xb,  n4seq, 1.0f);
  f32_to_bf16_kernel<<<(n4seq + 255) / 256, 256, 0, stream>>>(ctx, cb,  n4seq, 1.0f);
  // fold attention scale into Wq: (x @ (Wq*s)) @ K^T == (x@Wq @ K^T) * s
  f32_to_bf16_kernel<<<(n4w + 255) / 256,   256, 0, stream>>>(Wq,  wqb, n4w, ATT_SCALE);
  f32_to_bf16_kernel<<<(n4w + 255) / 256,   256, 0, stream>>>(Wk,  wkb, n4w, 1.0f);
  f32_to_bf16_kernel<<<(n4w + 255) / 256,   256, 0, stream>>>(Wv,  wvb, n4w, 1.0f);
  f32_to_bf16_kernel<<<(n4w + 255) / 256,   256, 0, stream>>>(Wo,  wob, n4w, 1.0f);

  const int Mrows = BATCH * SEQ_N;  // 4096
  dim3 blk(256);
  dim3 gproj(INNER / 128, Mrows / 128);  // (8, 32)
  gemm_bf16_128x128<unsigned short><<<gproj, blk, 0, stream>>>(xb, wqb, qb, Mrows, INNER, QDIM);
  gemm_bf16_128x128<unsigned short><<<gproj, blk, 0, stream>>>(cb, wkb, kb, Mrows, INNER, QDIM);
  gemm_bf16_128x128<unsigned short><<<gproj, blk, 0, stream>>>(cb, wvb, vb, Mrows, INNER, QDIM);

  flash_attn_kernel<<<BATCH * HEADS * (SEQ_N / 128), 256, 0, stream>>>(qb, kb, vb, ab);

  dim3 gout(QDIM / 128, Mrows / 128);
  gemm_bf16_128x128<float><<<gout, blk, 0, stream>>>(ab, wob, out, Mrows, QDIM, INNER);
}
